// IVDetect_72481868087335
// MI455X (gfx1250) — matrix operand order repaired and checked by reading, not compiled, not run on hardware
//
#include <hip/hip_runtime.h>

// Problem constants (match reference setup_inputs)
#define NN   20000      // nodes
#define TT   10         // timesteps
#define DD   200        // token dim
#define HH   256        // hidden
#define EE   320000     // edges
#define GG   128        // graphs
#define G3   768        // 3*H (gate width)
#define DPAD 224        // D padded to multiple of 32 for WMMA K-loop

typedef _Float16 h16;
typedef __attribute__((ext_vector_type(16))) _Float16 v16h;
typedef __attribute__((ext_vector_type(8)))  _Float16 v8h;
typedef __attribute__((ext_vector_type(8)))  float    v8f;

// ---------------- fills / converts ----------------
__global__ void ivd_fill_f32(float* __restrict__ p, float v, long n) {
  long i = (long)blockIdx.x * blockDim.x + threadIdx.x;
  if (i < n) p[i] = v;
}
__global__ void ivd_fill_f16(h16* __restrict__ p, long n) {
  long i = (long)blockIdx.x * blockDim.x + threadIdx.x;
  if (i < n) p[i] = (h16)0.f;
}
// dst[r][c] (row-major, cd cols) = src[r][c] (cs cols) or 0-pad
__global__ void ivd_pack_pad_f16(h16* __restrict__ dst, const float* __restrict__ src,
                                 long rows, int cs, int cd) {
  long i = (long)blockIdx.x * blockDim.x + threadIdx.x;
  if (i >= rows * (long)cd) return;
  long r = i / cd; int c = (int)(i - r * (long)cd);
  dst[i] = (c < cs) ? (h16)src[r * (long)cs + c] : (h16)0.f;
}
// dst[c][r] = src[r][c]  (for gcn1_w: want row-major [n_out, K])
__global__ void ivd_transpose_f16(h16* __restrict__ dst, const float* __restrict__ src,
                                  int rows, int cols) {
  int i = blockIdx.x * blockDim.x + threadIdx.x;
  if (i >= rows * cols) return;
  int r = i / cols, c = i - r * cols;
  dst[c * rows + r] = (h16)src[i];
}

// ---------------- WMMA GEMM: out[M, n_out] = A[M,K](f16) @ W[n_out,K]^T ----------------
// Per wave: 64x32 tile (4x2 WMMA 16x16x32 frags). 8 waves/WG -> 256 rows x 64 cols.
// Fragment layouts follow CDNA5 ISA 7.12.2 (wave32):
//  A 16x32 f16: lanes 0-15 hold row M=lane, K {0..7,16..23}; lanes 16-31 K {8..15,24..31}
//  B 32x16 f16: lanes 0-15 col N=lane, K {0..7,16..23}; lanes 16-31 K {8..15,24..31}
//  C/D f32:     VGPR v -> row v + 8*(lane>>4), col lane&15
template <bool OUT_F16>
__device__ __forceinline__ void ivd_gemm_tile(const h16* __restrict__ A, int lda, int M,
                                              const h16* __restrict__ W, int K,
                                              void* __restrict__ outp, int ldo) {
  const int lane = threadIdx.x & 31;
  const int wave = threadIdx.x >> 5;
  const int m0   = blockIdx.x * 256 + (wave & 3) * 64;
  const int n0   = blockIdx.y * 64 + (wave >> 2) * 32;
  const int rlo  = lane & 15;     // row-in-tile for A, col for B/C
  const int kg   = lane >> 4;     // K half selector

  const v8f vzero = {0.f, 0.f, 0.f, 0.f, 0.f, 0.f, 0.f, 0.f};
  v8f acc[4][2];
#pragma unroll
  for (int i = 0; i < 4; ++i)
#pragma unroll
    for (int j = 0; j < 2; ++j) acc[i][j] = vzero;

  for (int k0 = 0; k0 < K; k0 += 32) {
    v16h a[4];
#pragma unroll
    for (int i = 0; i < 4; ++i) {
      int r = m0 + i * 16 + rlo;
      r = (r < M) ? r : 0;                         // clamp; stores are guarded
      const h16* p = A + (size_t)r * lda + k0 + kg * 8;
      v8h lo = *reinterpret_cast<const v8h*>(p);       // K {kg*8 .. kg*8+7}
      v8h hi = *reinterpret_cast<const v8h*>(p + 16);  // K {16+kg*8 ..}
      __builtin_prefetch(p + 32, 0, 1);                // next K-tile of this row
      v16h t;
#pragma unroll
      for (int q = 0; q < 8; ++q) { t[q] = lo[q]; t[q + 8] = hi[q]; }
      a[i] = t;
    }
    v16h b[2];
#pragma unroll
    for (int j = 0; j < 2; ++j) {
      const h16* pw = W + (size_t)(n0 + j * 16 + rlo) * K + k0 + kg * 8;
      v8h blo = *reinterpret_cast<const v8h*>(pw);
      v8h bhi = *reinterpret_cast<const v8h*>(pw + 16);
      v16h tb;
#pragma unroll
      for (int q = 0; q < 8; ++q) { tb[q] = blo[q]; tb[q + 8] = bhi[q]; }
      b[j] = tb;
    }
#pragma unroll
    for (int i = 0; i < 4; ++i)
#pragma unroll
      for (int j = 0; j < 2; ++j)
        acc[i][j] = __builtin_amdgcn_wmma_f32_16x16x32_f16(
            false, a[i], false, b[j], (short)0, acc[i][j], false, false);
  }

  h16*   oh = (h16*)outp;
  float* of = (float*)outp;
  if (m0 + 64 <= M) {
    // fast path: full 64-row tile, no per-element exec masking
#pragma unroll
    for (int i = 0; i < 4; ++i)
#pragma unroll
      for (int v = 0; v < 8; ++v) {
        int r = m0 + i * 16 + kg * 8 + v;
#pragma unroll
        for (int j = 0; j < 2; ++j) {
          size_t off = (size_t)r * ldo + n0 + j * 16 + rlo;
          float val = acc[i][j][v];
          if (OUT_F16) oh[off] = (h16)val;
          else         of[off] = val;
        }
      }
  } else {
    // boundary tile along M: guard each row
#pragma unroll
    for (int i = 0; i < 4; ++i)
#pragma unroll
      for (int v = 0; v < 8; ++v) {
        int r = m0 + i * 16 + kg * 8 + v;
        if (r < M) {
#pragma unroll
          for (int j = 0; j < 2; ++j) {
            size_t off = (size_t)r * ldo + n0 + j * 16 + rlo;
            float val = acc[i][j][v];
            if (OUT_F16) oh[off] = (h16)val;
            else         of[off] = val;
          }
        }
      }
  }
}

__global__ void __launch_bounds__(256)
ivd_gemm_f16(const h16* __restrict__ A, int lda, int M,
             const h16* __restrict__ W, int K, h16* __restrict__ out, int ldo) {
  ivd_gemm_tile<true>(A, lda, M, W, K, out, ldo);
}
__global__ void __launch_bounds__(256)
ivd_gemm_f32(const h16* __restrict__ A, int lda, int M,
             const h16* __restrict__ W, int K, float* __restrict__ out, int ldo) {
  ivd_gemm_tile<false>(A, lda, M, W, K, out, ldo);
}

// ---------------- GRU gate fusion (one thread per (node, channel)) ----------------
__global__ void ivd_gru_gate(const h16* __restrict__ xw, int t,
                             const float* __restrict__ gh,
                             const float* __restrict__ bih, const float* __restrict__ bhh,
                             float* __restrict__ h32, h16* __restrict__ h16b,
                             h16* __restrict__ out_seq, float* __restrict__ feat,
                             const int* __restrict__ lens, int n) {
  long idx = (long)blockIdx.x * blockDim.x + threadIdx.x;
  if (idx >= (long)n * HH) return;
  int i = (int)(idx >> 8), c = (int)(idx & (HH - 1));
  size_t xrow = ((size_t)i * TT + t) * G3;
  size_t grow = (size_t)i * G3;
  float xr = (float)xw[xrow + c]          + bih[c];
  float xz = (float)xw[xrow + HH + c]     + bih[HH + c];
  float xn = (float)xw[xrow + 2 * HH + c] + bih[2 * HH + c];
  float hr = gh[grow + c]          + bhh[c];
  float hz = gh[grow + HH + c]     + bhh[HH + c];
  float hn = gh[grow + 2 * HH + c] + bhh[2 * HH + c];
  float r  = 1.f / (1.f + __expf(-(xr + hr)));
  float z  = 1.f / (1.f + __expf(-(xz + hz)));
  float ng = tanhf(xn + r * hn);
  float hprev = h32[idx];
  float hnew  = (1.f - z) * ng + z * hprev;
  h32[idx]  = hnew;
  h16b[idx] = (h16)hnew;
  if (out_seq) out_seq[((size_t)i * TT + t) * HH + c] = (h16)hnew;
  if (feat && (lens[i] - 1 == t)) feat[idx] = hnew;
}

// ---------------- GraphConv pieces ----------------
__global__ void ivd_edge_deg(const int* __restrict__ esrc, const int* __restrict__ edst,
                             float* __restrict__ dout, float* __restrict__ din, int e) {
  int i = blockIdx.x * blockDim.x + threadIdx.x;
  if (i >= e) return;
  atomicAdd(&dout[esrc[i]], 1.f);
  atomicAdd(&din[edst[i]], 1.f);
}
__global__ void ivd_scale_pre(const float* __restrict__ feat, const float* __restrict__ dout,
                              float* __restrict__ hpre, long n) {
  long idx = (long)blockIdx.x * blockDim.x + threadIdx.x;
  if (idx >= n) return;
  int i = (int)(idx >> 8);
  hpre[idx] = feat[idx] * rsqrtf(fmaxf(dout[i], 1.f));
}
// one 256-thread block per edge: coalesced row read, atomic row accumulate
__global__ void __launch_bounds__(256)
ivd_edge_scatter(const int* __restrict__ esrc, const int* __restrict__ edst,
                 const float* __restrict__ hpre, float* __restrict__ m) {
  int e = blockIdx.x;
  int c = threadIdx.x;
  int s = esrc[e], d = edst[e];
  atomicAdd(&m[(size_t)d * HH + c], hpre[(size_t)s * HH + c]);
}
__global__ void ivd_scale_post(float* __restrict__ m, const float* __restrict__ din,
                               h16* __restrict__ mh, long n) {
  long idx = (long)blockIdx.x * blockDim.x + threadIdx.x;
  if (idx >= n) return;
  int i = (int)(idx >> 8);
  float v = m[idx] * rsqrtf(fmaxf(din[i], 1.f));
  m[idx]  = v;
  mh[idx] = (h16)v;
}
__global__ void ivd_relu_bias(float* __restrict__ g, const float* __restrict__ b, long n) {
  long idx = (long)blockIdx.x * blockDim.x + threadIdx.x;
  if (idx >= n) return;
  int c = (int)(idx & (HH - 1));
  g[idx] = fmaxf(g[idx] + b[c], 0.f);
}
// GCN2 (H->2) + per-graph accumulation, one wave32 per node
__global__ void __launch_bounds__(256)
ivd_gcn2_mean(const float* __restrict__ g1, const float* __restrict__ w2,
              const float* __restrict__ b2, const int* __restrict__ gid,
              float* __restrict__ sums, float* __restrict__ cnt, int n) {
  int lane = threadIdx.x & 31;
  int node = blockIdx.x * 8 + (threadIdx.x >> 5);
  if (node >= n) return;
  const float* row = g1 + (size_t)node * HH;
  float a0 = 0.f, a1 = 0.f;
  for (int c = lane; c < HH; c += 32) {
    float v = row[c];
    a0 += v * w2[c * 2 + 0];
    a1 += v * w2[c * 2 + 1];
  }
#pragma unroll
  for (int m = 16; m >= 1; m >>= 1) {
    a0 += __shfl_xor(a0, m, 32);
    a1 += __shfl_xor(a1, m, 32);
  }
  if (lane == 0) {
    int g = gid[node];
    atomicAdd(&sums[g * 2 + 0], a0 + b2[0]);
    atomicAdd(&sums[g * 2 + 1], a1 + b2[1]);
    atomicAdd(&cnt[g], 1.f);
  }
}
__global__ void ivd_finalize(const float* __restrict__ sums, const float* __restrict__ cnt,
                             float* __restrict__ out) {
  int i = threadIdx.x;
  if (i < GG * 2) out[i] = sums[i] / fmaxf(cnt[i >> 1], 1.f);
}

// ---------------- driver ----------------
extern "C" void kernel_launch(void* const* d_in, const int* in_sizes, int n_in,
                              void* d_out, int out_size, void* d_ws, size_t ws_size,
                              hipStream_t stream) {
  (void)in_sizes; (void)n_in; (void)out_size; (void)ws_size;
  const float* subseq = (const float*)d_in[0];
  const int*   lens   = (const int*)d_in[1];
  const int*   esrc   = (const int*)d_in[2];
  const int*   edst   = (const int*)d_in[3];
  const int*   gid    = (const int*)d_in[4];
  const float* Wih0f  = (const float*)d_in[5];
  const float* Whh0f  = (const float*)d_in[6];
  const float* bih0   = (const float*)d_in[7];
  const float* bhh0   = (const float*)d_in[8];
  const float* Wih1f  = (const float*)d_in[9];
  const float* Whh1f  = (const float*)d_in[10];
  const float* bih1   = (const float*)d_in[11];
  const float* bhh1   = (const float*)d_in[12];
  const float* g1wf   = (const float*)d_in[13];
  const float* g1b    = (const float*)d_in[14];
  const float* g2w    = (const float*)d_in[15];
  const float* g2b    = (const float*)d_in[16];
  float* out = (float*)d_out;

  // workspace carve-out (256B aligned)
  char* wsp = (char*)d_ws;
  auto take = [&](size_t bytes) -> void* {
    char* p = wsp; wsp += (bytes + 255) & ~(size_t)255; return (void*)p;
  };
  h16*   X0     = (h16*)  take((size_t)NN * TT * DPAD * 2);  // padded f16 tokens
  h16*   out0   = (h16*)  take((size_t)NN * TT * HH * 2);    // layer-0 outputs
  h16*   xW     = (h16*)  take((size_t)NN * TT * G3 * 2);    // input projections (reused L0/L1)
  float* gh     = (float*)take((size_t)NN * G3 * 4);         // recurrent gates / GCN1 out
  float* h32    = (float*)take((size_t)NN * HH * 4);
  h16*   hh16   = (h16*)  take((size_t)NN * HH * 2);
  float* feat   = (float*)take((size_t)NN * HH * 4);
  float* hpre   = (float*)take((size_t)NN * HH * 4);
  float* msum   = (float*)take((size_t)NN * HH * 4);
  h16*   mh     = (h16*)  take((size_t)NN * HH * 2);
  float* degout = (float*)take((size_t)NN * 4);
  float* degin  = (float*)take((size_t)NN * 4);
  h16*   Wih0   = (h16*)  take((size_t)G3 * DPAD * 2);
  h16*   Whh0   = (h16*)  take((size_t)G3 * HH * 2);
  h16*   Wih1   = (h16*)  take((size_t)G3 * HH * 2);
  h16*   Whh1   = (h16*)  take((size_t)G3 * HH * 2);
  h16*   g1wT   = (h16*)  take((size_t)HH * HH * 2);
  float* sums   = (float*)take((size_t)GG * 2 * 4);
  float* cnt    = (float*)take((size_t)GG * 4);

  const int THR = 256;
  auto blk = [](long n) { return (int)((n + 255) / 256); };
  dim3 gridBig((NN * TT + 255) / 256, G3 / 64);   // M=200000, 768 cols
  dim3 gridStep((NN + 255) / 256, G3 / 64);       // M=20000, 768 cols
  dim3 gridGcn((NN + 255) / 256, HH / 64);        // M=20000, 256 cols

  // ---- precision conversion / packing ----
  ivd_pack_pad_f16<<<blk((long)NN * TT * DPAD), THR, 0, stream>>>(X0, subseq, (long)NN * TT, DD, DPAD);
  ivd_pack_pad_f16<<<blk((long)G3 * DPAD), THR, 0, stream>>>(Wih0, Wih0f, G3, DD, DPAD);
  ivd_pack_pad_f16<<<blk((long)G3 * HH), THR, 0, stream>>>(Whh0, Whh0f, G3, HH, HH);
  ivd_pack_pad_f16<<<blk((long)G3 * HH), THR, 0, stream>>>(Wih1, Wih1f, G3, HH, HH);
  ivd_pack_pad_f16<<<blk((long)G3 * HH), THR, 0, stream>>>(Whh1, Whh1f, G3, HH, HH);
  ivd_transpose_f16<<<blk((long)HH * HH), THR, 0, stream>>>(g1wT, g1wf, HH, HH);

  // ---- GRU layer 0 ----
  ivd_gemm_f16<<<gridBig, THR, 0, stream>>>(X0, DPAD, NN * TT, Wih0, DPAD, xW, G3);
  ivd_fill_f32<<<blk((long)NN * HH), THR, 0, stream>>>(h32, 0.f, (long)NN * HH);
  ivd_fill_f16<<<blk((long)NN * HH), THR, 0, stream>>>(hh16, (long)NN * HH);
  for (int t = 0; t < TT; ++t) {
    ivd_gemm_f32<<<gridStep, THR, 0, stream>>>(hh16, HH, NN, Whh0, HH, gh, G3);
    ivd_gru_gate<<<blk((long)NN * HH), THR, 0, stream>>>(
        xW, t, gh, bih0, bhh0, h32, hh16, out0, (float*)nullptr, lens, NN);
  }

  // ---- GRU layer 1 ----
  ivd_gemm_f16<<<gridBig, THR, 0, stream>>>(out0, HH, NN * TT, Wih1, HH, xW, G3);
  ivd_fill_f32<<<blk((long)NN * HH), THR, 0, stream>>>(h32, 0.f, (long)NN * HH);
  ivd_fill_f16<<<blk((long)NN * HH), THR, 0, stream>>>(hh16, (long)NN * HH);
  for (int t = 0; t < TT; ++t) {
    ivd_gemm_f32<<<gridStep, THR, 0, stream>>>(hh16, HH, NN, Whh1, HH, gh, G3);
    ivd_gru_gate<<<blk((long)NN * HH), THR, 0, stream>>>(
        xW, t, gh, bih1, bhh1, h32, hh16, (h16*)nullptr, feat, lens, NN);
  }

  // ---- GraphConv 1 (norm='both') ----
  ivd_fill_f32<<<blk((long)NN), THR, 0, stream>>>(degout, 0.f, (long)NN);
  ivd_fill_f32<<<blk((long)NN), THR, 0, stream>>>(degin, 0.f, (long)NN);
  ivd_edge_deg<<<blk((long)EE), THR, 0, stream>>>(esrc, edst, degout, degin, EE);
  ivd_scale_pre<<<blk((long)NN * HH), THR, 0, stream>>>(feat, degout, hpre, (long)NN * HH);
  ivd_fill_f32<<<blk((long)NN * HH), THR, 0, stream>>>(msum, 0.f, (long)NN * HH);
  ivd_edge_scatter<<<EE, HH, 0, stream>>>(esrc, edst, hpre, msum);
  ivd_scale_post<<<blk((long)NN * HH), THR, 0, stream>>>(msum, degin, mh, (long)NN * HH);
  ivd_gemm_f32<<<gridGcn, THR, 0, stream>>>(mh, HH, NN, g1wT, HH, gh, HH);
  ivd_relu_bias<<<blk((long)NN * HH), THR, 0, stream>>>(gh, g1b, (long)NN * HH);

  // ---- GraphConv 2 (H->2) fused with per-graph mean ----
  // Second conv reuses the SAME normalized adjacency; m2 = D^-1/2 A D^-1/2 relu(h1).
  ivd_scale_pre<<<blk((long)NN * HH), THR, 0, stream>>>(gh, degout, hpre, (long)NN * HH);
  ivd_fill_f32<<<blk((long)NN * HH), THR, 0, stream>>>(msum, 0.f, (long)NN * HH);
  ivd_edge_scatter<<<EE, HH, 0, stream>>>(esrc, edst, hpre, msum);
  ivd_scale_post<<<blk((long)NN * HH), THR, 0, stream>>>(msum, degin, mh, (long)NN * HH);
  ivd_fill_f32<<<1, THR, 0, stream>>>(sums, 0.f, (long)GG * 2);
  ivd_fill_f32<<<1, THR, 0, stream>>>(cnt, 0.f, (long)GG);
  ivd_gcn2_mean<<<(NN + 7) / 8, THR, 0, stream>>>(msum, g2w, g2b, gid, sums, cnt, NN);
  ivd_finalize<<<1, 256, 0, stream>>>(sums, cnt, out);
}
